// WSISS_72155450573007
// MI455X (gfx1250) — matrix operand order repaired and checked by reading, not compiled, not run on hardware
//
#include <hip/hip_runtime.h>
#include <hip/hip_bf16.h>

#define EPSF 1e-5f

typedef __attribute__((ext_vector_type(16))) __bf16 v16bf;
typedef __attribute__((ext_vector_type(8)))  float  v8f;
typedef __attribute__((ext_vector_type(4)))  unsigned tg0_t;
typedef __attribute__((ext_vector_type(8)))  int      tg1_t;
typedef __attribute__((ext_vector_type(4)))  int      tg2_t;

union V16 { uint4 q[2]; unsigned u[8]; v16bf v; };

__device__ __forceinline__ unsigned short f2bf(float f) {
    unsigned u = __float_as_uint(f);
    unsigned r = u + 0x7FFFu + ((u >> 16) & 1u);   // round-to-nearest-even
    return (unsigned short)(r >> 16);
}
// packed f32x2 -> bf16x2 in one VALU op (inputs are plain VALU results,
// never WMMA destination registers, so no hidden WMMA->asm hazard)
__device__ __forceinline__ unsigned packbf(float a, float b) {
    unsigned r;
    asm("v_cvt_pk_bf16_f32 %0, %1, %2" : "=v"(r) : "v"(a), "v"(b));
    return r;
}
// single v_max_num_f32 relu; select form folds to max_num (NaN -> 0 matches),
// and as IR the backend inserts the WMMA->VALU hazard NOPs correctly.
__device__ __forceinline__ float relu1(float x) {
    return x > 0.f ? x : 0.f;
}

// contiguous 16 bf16 (B-tile style)
__device__ __forceinline__ v16bf ldB(const unsigned short* p) {
    V16 t; t.q[0] = *(const uint4*)p; t.q[1] = *(const uint4*)(p + 8); return t.v;
}
// contiguous 16 bf16 (A-tile style)
__device__ __forceinline__ v16bf ldA(const unsigned short* p) {
    V16 t; t.q[0] = *(const uint4*)p; t.q[1] = *(const uint4*)(p + 8); return t.v;
}

#define WMMA_BF16(A, B, C) \
    __builtin_amdgcn_wmma_f32_16x16x32_bf16(false, (A), false, (B), (short)0, (C), false, false)

// Tensor Data Mover: 2D tile (data_size = 2 bytes) global -> LDS
__device__ __forceinline__ void tdm_load_2d(unsigned lds_off, unsigned long long gaddr,
                                            unsigned tensor_d0, unsigned tensor_d1,
                                            unsigned tile_d0, unsigned tile_d1,
                                            unsigned long long stride0) {
    tg0_t g0;
    g0.x = 1u;                                           // count=1, user descriptor
    g0.y = lds_off;                                      // lds_addr (bytes)
    g0.z = (unsigned)gaddr;                              // global_addr[31:0]
    g0.w = ((unsigned)(gaddr >> 32) & 0x01FFFFFFu) | 0x80000000u; // addr[56:32], type=2
    tg1_t g1;
    g1[0] = 0x00010000;                                  // wg_mask=0, data_size=1 (2B)
    g1[1] = (int)((tensor_d0 & 0xFFFFu) << 16);          // tensor_dim0[15:0]
    g1[2] = (int)((tensor_d0 >> 16) | ((tensor_d1 & 0xFFFFu) << 16));
    g1[3] = (int)((tensor_d1 >> 16) | (tile_d0 << 16));  // tile_dim0
    g1[4] = (int)(tile_d1 & 0xFFFFu);                    // tile_dim1, tile_dim2=0
    g1[5] = (int)(unsigned)(stride0 & 0xFFFFFFFFu);      // tensor_dim0_stride[31:0]
    g1[6] = (int)(unsigned)(stride0 >> 32);              // stride0[47:32], stride1[15:0]=0
    g1[7] = 0;
    tg2_t z4 = {0, 0, 0, 0};
    tg1_t z8 = {0, 0, 0, 0, 0, 0, 0, 0};
    __builtin_amdgcn_tensor_load_to_lds(g0, g1, z4, z4, z8, 0);
}

__device__ __forceinline__ float bilin_ac(const float* __restrict__ p, int H, int W,
                                          float sy, float sx, int y, int x) {
    float yi = y * sy, xi = x * sx;
    int y0 = (int)floorf(yi); y0 = min(max(y0, 0), H - 2);
    int x0 = (int)floorf(xi); x0 = min(max(x0, 0), W - 2);
    float wy = yi - (float)y0, wx = xi - (float)x0;
    const float* r0 = p + y0 * W + x0;
    const float* r1 = r0 + W;
    float a = r0[0], b = r0[1], c = r1[0], d = r1[1];
    float top = a + (b - a) * wx, bot = c + (d - c) * wx;
    return top + (bot - top) * wy;
}

// ---------- 1) cam = W @ relu(d_conv1), 64x4096 ----------
__global__ __launch_bounds__(256) void k_cam(const float* __restrict__ dconv,
                                             const float* __restrict__ W,
                                             float* __restrict__ cam) {
    int p = blockIdx.x * 256 + threadIdx.x;
    int o = blockIdx.y;
    float s = 0.f;
#pragma unroll 8
    for (int c = 0; c < 64; ++c)
        s += fmaxf(dconv[c * 4096 + p], 0.f) * W[o * 64 + c];
    cam[o * 4096 + p] = s;
}

// ---------- 2) per-channel max of relu(cam) + EPS ----------
__global__ __launch_bounds__(256) void k_chanmax(const float* __restrict__ cam,
                                                 float* __restrict__ cmax) {
    __shared__ float red[256];
    int c = blockIdx.x;
    float m = -3.4e38f;
    for (int i = threadIdx.x; i < 4096; i += 256) m = fmaxf(m, cam[c * 4096 + i]);
    red[threadIdx.x] = m; __syncthreads();
    for (int s = 128; s > 0; s >>= 1) {
        if (threadIdx.x < s) red[threadIdx.x] = fmaxf(red[threadIdx.x], red[threadIdx.x + s]);
        __syncthreads();
    }
    if (threadIdx.x == 0) cmax[c] = fmaxf(red[0], 0.f) + EPSF;
}

// ---------- 3) cam_d_norm with bg/fg logic ----------
__global__ __launch_bounds__(256) void k_fgbg(const float* __restrict__ cam,
                                              const float* __restrict__ cmax,
                                              float* __restrict__ cdn) {
    int p = blockIdx.x * 256 + threadIdx.x;
    float m = 0.f;
    for (int c = 1; c < 64; ++c) {
        float v = fmaxf(cam[c * 4096 + p] - EPSF, 0.f) / cmax[c];
        m = fmaxf(m, v);
    }
    cdn[p] = 1.f - m;
    for (int c = 1; c < 64; ++c) {
        float v = fmaxf(cam[c * 4096 + p] - EPSF, 0.f) / cmax[c];
        cdn[c * 4096 + p] = (v < m) ? 0.f : v;
    }
}

// ---------- 4) f = W @ img, 64x65536 ----------
__global__ __launch_bounds__(256) void k_f(const float* __restrict__ img,
                                           const float* __restrict__ W,
                                           float* __restrict__ f) {
    int p = blockIdx.x * 256 + threadIdx.x;
    int o = blockIdx.y;
    float s = 0.f;
#pragma unroll 8
    for (int c = 0; c < 64; ++c)
        s += img[c * 65536 + p] * W[o * 64 + c];
    f[o * 65536 + p] = s;
}

// ---------- 5) resize cam_d_norm 64->128, to bf16 [c][16384] ----------
__global__ __launch_bounds__(256) void k_camv(const float* __restrict__ cdn,
                                              unsigned short* __restrict__ camv) {
    int idx = blockIdx.x * 256 + threadIdx.x;
    int c = idx >> 14, j = idx & 16383, y = j >> 7, x = j & 127;
    float v = bilin_ac(cdn + c * 4096, 64, 64, 63.f / 127.f, 63.f / 127.f, y, x);
    camv[idx] = f2bf(v);
}

// ---------- 6) resize f 256->128 (f32) ----------
__global__ __launch_bounds__(256) void k_fi(const float* __restrict__ f,
                                            float* __restrict__ fi) {
    int idx = blockIdx.x * 256 + threadIdx.x;
    int c = idx >> 14, j = idx & 16383, y = j >> 7, x = j & 127;
    fi[idx] = bilin_ac(f + c * 65536, 256, 256, 255.f / 127.f, 255.f / 127.f, y, x);
}

// ---------- 7) per-position L2 normalize, transpose to bf16 [pos][64] ----------
__global__ __launch_bounds__(256) void k_fvT(const float* __restrict__ fi,
                                             unsigned short* __restrict__ fvT) {
    int j = blockIdx.x * 256 + threadIdx.x;
    float s = 0.f;
#pragma unroll 8
    for (int c = 0; c < 64; ++c) { float v = fi[c * 16384 + j]; s += v * v; }
    float inv = 1.f / (sqrtf(s) + EPSF);
#pragma unroll 8
    for (int c = 0; c < 64; ++c) fvT[j * 64 + c] = f2bf(fi[c * 16384 + j] * inv);
}

// ---------- 8) fused: S = relu(fv^T fv), colsum, cam_rv = camv @ S / colsum ----------
// Block = 4 waves, 64 j columns. Shared i-chunk (64 rows) staged in LDS by the
// Tensor Data Mover, double-buffered: wave0 DMAs the fvT tile, wave1 the camv tile.
#define CH 64
__global__ __launch_bounds__(128) void k_aff_fused(const unsigned short* __restrict__ fvT,
                                                   const unsigned short* __restrict__ camv,
                                                   float* __restrict__ camrv) {
    __shared__ unsigned short ls_f[2][CH * 64];   // [i-local][64 ch]
    __shared__ unsigned short ls_c[2][64 * CH];   // [c][i-local]

    const int lane = threadIdx.x & 31;
    const int wave = threadIdx.x >> 5;
    const int j0   = blockIdx.x * 64 + wave * 16;
    const int hf   = lane >> 4;
    const int l15  = lane & 15;
    const int kbase = hf * 16;           // A-layout K base
    const int koff  = hf * 16;           // B-layout K base
    const int NCH = 16384 / CH;

    // B tiles for this j-strip (channels 0..31 / 32..63), resident all loop
    const unsigned short* bp = fvT + (j0 + l15) * 64 + koff;
    v16bf Bk0 = ldB(bp);
    v16bf Bk1 = ldB(bp + 32);

    v8f acc0 = {}, acc1 = {}, acc2 = {}, acc3 = {};
    float cs0 = 0.f, cs1 = 0.f, cs2 = 0.f, cs3 = 0.f;   // split colsum accumulators

    // prologue: DMA chunk 0 into buffer 0
    if (wave == 0)
        tdm_load_2d((unsigned)(uintptr_t)&ls_f[0][0],
                    (unsigned long long)(uintptr_t)fvT,
                    64, 16384, 64, CH, 64);
    else if (wave == 1)
        tdm_load_2d((unsigned)(uintptr_t)&ls_c[0][0],
                    (unsigned long long)(uintptr_t)camv,
                    16384, 64, CH, 64, 16384);

    for (int ck = 0; ck < NCH; ++ck) {
        const int buf = ck & 1;
        if (wave < 2) {
            if (ck + 1 < NCH) {
                // issue next chunk into the other buffer, then wait for current
                if (wave == 0)
                    tdm_load_2d((unsigned)(uintptr_t)&ls_f[buf ^ 1][0],
                                (unsigned long long)(uintptr_t)(fvT + (ck + 1) * CH * 64),
                                64, 16384, 64, CH, 64);
                else
                    tdm_load_2d((unsigned)(uintptr_t)&ls_c[buf ^ 1][0],
                                (unsigned long long)(uintptr_t)(camv + (ck + 1) * CH),
                                16384, 64, CH, 64, 16384);
                __builtin_amdgcn_s_wait_tensorcnt(1);   // current chunk landed (in-order)
            } else {
                __builtin_amdgcn_s_wait_tensorcnt(0);
            }
        }
        __syncthreads();   // chunk ck visible to all waves

        const unsigned short* lf = &ls_f[buf][0];
        const unsigned short* lc = &ls_c[buf][0];

#pragma unroll
        for (int li = 0; li < CH; li += 32) {
            // ---- GEMM1: S (32 x 16) = fvT(i-block)^T x fv(j-strip), K = 64 ch ----
            const unsigned short* a0 = lf + (li + l15) * 64 + kbase;
            const unsigned short* a1 = lf + (li + 16 + l15) * 64 + kbase;
            v16bf A00 = ldA(a0), A01 = ldA(a0 + 32);
            v16bf A10 = ldA(a1), A11 = ldA(a1 + 32);
            v8f S0 = {}; S0 = WMMA_BF16(A00, Bk0, S0); S0 = WMMA_BF16(A01, Bk1, S0);
            v8f S1 = {}; S1 = WMMA_BF16(A10, Bk0, S1); S1 = WMMA_BF16(A11, Bk1, S1);

            // relu + column-sum accumulation (4 rotating partials, short chains)
#pragma unroll
            for (int r = 0; r < 8; ++r) {
                float s0 = relu1(S0[r]), s1 = relu1(S1[r]);
                S0[r] = s0; S1[r] = s1;
                switch (r & 3) {
                    case 0: cs0 += s0 + s1; break;
                    case 1: cs1 += s0 + s1; break;
                    case 2: cs2 += s0 + s1; break;
                    default: cs3 += s0 + s1; break;
                }
            }

            // ---- repack S (f32 D-layout) -> B2 (bf16 B-layout, K = i) ----
            V16 b2;
#pragma unroll
            for (int r = 0; r < 4; ++r) {
                unsigned x0  = packbf(S0[2 * r], S0[2 * r + 1]);
                unsigned x1  = packbf(S1[2 * r], S1[2 * r + 1]);
                unsigned x0x = __shfl_xor(x0, 16, 32);
                unsigned x1x = __shfl_xor(x1, 16, 32);
                b2.u[r]     = hf ? x1x : x0;
                b2.u[4 + r] = hf ? x1  : x0x;
            }
            v16bf B2 = b2.v;

            // ---- GEMM2: cam_rv(c-tile, j) += camv(c-tile, i-block) x S ----
            const unsigned short* c0 = lc + (0  + l15) * CH + li + kbase;
            const unsigned short* c1 = lc + (16 + l15) * CH + li + kbase;
            const unsigned short* c2 = lc + (32 + l15) * CH + li + kbase;
            const unsigned short* c3 = lc + (48 + l15) * CH + li + kbase;
            acc0 = WMMA_BF16(ldA(c0), B2, acc0);
            acc1 = WMMA_BF16(ldA(c1), B2, acc1);
            acc2 = WMMA_BF16(ldA(c2), B2, acc2);
            acc3 = WMMA_BF16(ldA(c3), B2, acc3);
        }
        __syncthreads();   // all waves done reading buf before it is re-filled
    }

    // full column sum (combine halves), late normalization of aff
    float csum  = (cs0 + cs1) + (cs2 + cs3);
    float ctot  = csum + __shfl_xor(csum, 16, 32);
    float scale = 1.f / (ctot + EPSF);

    int j = j0 + l15;
    int mrow = 8 * hf;
#pragma unroll
    for (int r = 0; r < 8; ++r) {
        camrv[(0  + r + mrow) * 16384 + j] = acc0[r] * scale;
        camrv[(16 + r + mrow) * 16384 + j] = acc1[r] * scale;
        camrv[(32 + r + mrow) * 16384 + j] = acc2[r] * scale;
        camrv[(48 + r + mrow) * 16384 + j] = acc3[r] * scale;
    }
}

// ---------- 9) upsample cam_rv 128->256 (256->256 resize is identity) ----------
__global__ __launch_bounds__(256) void k_up_rv(const float* __restrict__ camrv,
                                               float* __restrict__ out) {
    int idx = blockIdx.x * 256 + threadIdx.x;
    int c = idx >> 16, j = idx & 65535, y = j >> 8, x = j & 255;
    out[idx] = bilin_ac(camrv + c * 16384, 128, 128, 127.f / 255.f, 127.f / 255.f, y, x);
}

// ---------- 10) upsample cam 64->256 ----------
__global__ __launch_bounds__(256) void k_up_cam(const float* __restrict__ cam,
                                                float* __restrict__ out) {
    int idx = blockIdx.x * 256 + threadIdx.x;
    int c = idx >> 16, j = idx & 65535, y = j >> 8, x = j & 255;
    out[idx] = bilin_ac(cam + c * 4096, 64, 64, 63.f / 255.f, 63.f / 255.f, y, x);
}

extern "C" void kernel_launch(void* const* d_in, const int* in_sizes, int n_in,
                              void* d_out, int out_size, void* d_ws, size_t ws_size,
                              hipStream_t stream) {
    const float* dconv = (const float*)d_in[0];   // [1,64,64,64]
    const float* img   = (const float*)d_in[1];   // [1,64,256,256]
    const float* W     = (const float*)d_in[2];   // [64,64]
    float* out = (float*)d_out;                   // cam (64*256*256) then cam_rv

    float* cam   = (float*)d_ws;                  // 64*4096
    float* cmax  = cam   + 262144;                // 64 (padded 256)
    float* cdn   = cmax  + 256;                   // 64*4096
    float* f     = cdn   + 262144;                // 64*65536
    float* fi    = f     + 4194304;               // 64*16384
    float* camrv = fi    + 1048576;               // 64*16384
    unsigned short* camv = (unsigned short*)(camrv + 1048576); // 64*16384 bf16
    unsigned short* fvT  = camv + 1048576;                     // 16384*64 bf16

    k_cam    <<<dim3(16, 64),  256, 0, stream>>>(dconv, W, cam);
    k_chanmax<<<64,            256, 0, stream>>>(cam, cmax);
    k_fgbg   <<<16,            256, 0, stream>>>(cam, cmax, cdn);
    k_f      <<<dim3(256, 64), 256, 0, stream>>>(img, W, f);
    k_camv   <<<4096,          256, 0, stream>>>(cdn, camv);
    k_fi     <<<4096,          256, 0, stream>>>(f, fi);
    k_fvT    <<<64,            256, 0, stream>>>(fi, fvT);
    k_aff_fused<<<256,         128, 0, stream>>>(fvT, camv, camrv);
    k_up_cam <<<16384,         256, 0, stream>>>(cam, out);
    k_up_rv  <<<16384,         256, 0, stream>>>(camrv, out + 64 * 65536);
}
